// Qwen3Attention_5729486373588
// MI455X (gfx1250) — matrix operand order repaired and checked by reading, not compiled, not run on hardware
//
#include <hip/hip_runtime.h>
#include <hip/hip_bf16.h>
#include <stdint.h>

// ---------------- CDNA5 types / helpers ----------------
typedef _Float16 v16h __attribute__((ext_vector_type(16)));
typedef _Float16 v8h  __attribute__((ext_vector_type(8)));
typedef float    v8f  __attribute__((ext_vector_type(8)));

union AF { v16h v; v8h h[2]; };

__device__ __forceinline__ v8f wmma16(v16h a, v16h b, v8f c) {
  // D = A(16x32 f16) x B(32x16 f16) + C(16x16 f32)
  return __builtin_amdgcn_wmma_f32_16x16x32_f16(false, a, false, b, (short)0, c,
                                                false, false);
}

// Low 32 bits of a flat LDS pointer == LDS byte offset (flat aperture maps
// addr[31:0] directly to LDS; SHARED_BASE lives in addr[63:32]).
__device__ __forceinline__ unsigned ldsOff(const void* p) {
  return (unsigned)(uintptr_t)p;
}

__device__ __forceinline__ void asyncLoadB128(unsigned lds, const void* g) {
  asm volatile("global_load_async_to_lds_b128 %0, %1, off"
               :: "v"(lds), "v"(g) : "memory");
}
__device__ __forceinline__ void waitAsync() {
  asm volatile("s_wait_asynccnt 0x0" ::: "memory");
}

// ---------------- precision converts ----------------
__global__ void conv1d(const float* __restrict__ src, _Float16* __restrict__ dst,
                       size_t n) {
  size_t i = (size_t)blockIdx.x * blockDim.x + threadIdx.x;
  size_t stride = (size_t)gridDim.x * blockDim.x;
  for (; i < n; i += stride) dst[i] = (_Float16)src[i];
}

__global__ void conv2d(const float* __restrict__ src, _Float16* __restrict__ dst,
                       int rows, int cols, int dstStride, int colOff) {
  size_t n = (size_t)rows * cols;
  size_t i = (size_t)blockIdx.x * blockDim.x + threadIdx.x;
  size_t stride = (size_t)gridDim.x * blockDim.x;
  for (; i < n; i += stride) {
    size_t r = i / (size_t)cols, c = i % (size_t)cols;
    dst[r * (size_t)dstStride + (size_t)colOff + c] = (_Float16)src[i];
  }
}

// ---------------- WMMA GEMM: C(f32 MxN) = A(f16 MxK) * B(f16 KxN) ----------------
// Block tile 128x128, BK=32, 8 waves (2x4), wave tile 64x32 (4x2 WMMA frags).
__global__ __launch_bounds__(256) void gemm_wmma(const _Float16* __restrict__ A,
                                                 const _Float16* __restrict__ B,
                                                 float* __restrict__ C,
                                                 int M, int N, int K) {
  __shared__ __align__(16) _Float16 sA[128 * 32];   // [m][k]
  __shared__ __align__(16) _Float16 sB[128 * 32];   // [n][k] (transposed)
  const int tid  = threadIdx.x;
  const int lane = tid & 31, wave = tid >> 5;
  const int ln   = lane & 15, hf = lane >> 4;
  const int wm   = wave >> 2, wn = wave & 3;         // 2 x 4 wave grid
  const int bm0  = blockIdx.y * 128, bn0 = blockIdx.x * 128;

  v8f acc[4][2];
#pragma unroll
  for (int mi = 0; mi < 4; mi++)
#pragma unroll
    for (int ni = 0; ni < 2; ni++)
#pragma unroll
      for (int r = 0; r < 8; r++) acc[mi][ni][r] = 0.0f;

  // per-thread staging assignments
  const int am = tid >> 1, ak = (tid & 1) * 16;      // sA: 16 f16 per thread
  const unsigned saOff = ldsOff(&sA[am * 32 + ak]);
  const _Float16* ag = A + (size_t)(bm0 + am) * K + ak;
  const int bk = tid >> 3, bn = (tid & 7) * 16;      // sB: 16 f16 per thread
  const _Float16* bg = B + (size_t)bk * N + bn0 + bn;

  for (int kk = 0; kk < K; kk += 32) {
    // A tile: straight async copy global -> LDS (rows contiguous in k)
    asyncLoadB128(saOff,      ag + kk);
    asyncLoadB128(saOff + 16, ag + kk + 8);
    // B tile: global b128 load + transposed ds stores -> sB[n][k]
    v8h b0 = *(const v8h*)(bg + (size_t)kk * N);
    v8h b1 = *(const v8h*)(bg + (size_t)kk * N + 8);
#pragma unroll
    for (int i = 0; i < 8; i++) {
      sB[(bn + i) * 32 + bk]     = b0[i];
      sB[(bn + 8 + i) * 32 + bk] = b1[i];
    }
    if (kk + 32 < K) {
      __builtin_prefetch(ag + kk + 32, 0, 1);
      __builtin_prefetch(bg + (size_t)(kk + 32) * N, 0, 1);
    }
    waitAsync();
    __syncthreads();

    AF af[4], bf[2];
#pragma unroll
    for (int mi = 0; mi < 4; mi++) {   // A frag: lane=row(M), elems->K pattern
      const _Float16* p = &sA[(wm * 64 + mi * 16 + ln) * 32 + hf * 8];
      af[mi].h[0] = *(const v8h*)p;
      af[mi].h[1] = *(const v8h*)(p + 16);
    }
#pragma unroll
    for (int ni = 0; ni < 2; ni++) {   // B frag: lane=col(N), elems->K pattern
      const _Float16* p = &sB[(wn * 32 + ni * 16 + ln) * 32 + hf * 8];
      bf[ni].h[0] = *(const v8h*)p;
      bf[ni].h[1] = *(const v8h*)(p + 16);
    }
#pragma unroll
    for (int mi = 0; mi < 4; mi++)
#pragma unroll
      for (int ni = 0; ni < 2; ni++)
        acc[mi][ni] = wmma16(af[mi].v, bf[ni].v, acc[mi][ni]);
    __syncthreads();
  }

#pragma unroll
  for (int mi = 0; mi < 4; mi++)
#pragma unroll
    for (int ni = 0; ni < 2; ni++)
#pragma unroll
      for (int r = 0; r < 8; r++) {
        const int gm = bm0 + wm * 64 + mi * 16 + r + 8 * hf;
        const int gn = bn0 + wn * 32 + ni * 16 + ln;
        C[(size_t)gm * N + gn] = acc[mi][ni][r];
      }
}

// ---------------- RMSNorm + interleaved RoPE + V transpose ----------------
// grid: (48 slots, 4096 tokens); slots 0..31 -> Q heads, 32..39 -> K heads,
// 40..47 -> V heads. 32 threads, 4 dims each (2 interleaved RoPE pairs).
__global__ __launch_bounds__(32) void prep(const float* __restrict__ qkv,
                                           const float* __restrict__ qw,
                                           const float* __restrict__ kw,
                                           _Float16* __restrict__ qo,
                                           _Float16* __restrict__ ko,
                                           _Float16* __restrict__ vt) {
  const int slot = blockIdx.x;
  const int token = blockIdx.y;                 // 0..4095 = b*s
  const int si = token & 2047, bi = token >> 11;
  const int t = threadIdx.x;
  const float* row = qkv + (size_t)token * 6144;

  int kind, hh, col;
  if (slot < 32)      { kind = 0; hh = slot;      col = hh * 128; }
  else if (slot < 40) { kind = 1; hh = slot - 32; col = 4096 + hh * 128; }
  else                { kind = 2; hh = slot - 40; col = 5120 + hh * 128; }

  const int d0 = t * 4;
  float x0 = row[col + d0], x1 = row[col + d0 + 1];
  float x2 = row[col + d0 + 2], x3 = row[col + d0 + 3];

  if (kind == 2) {  // V: no norm/rope, store transposed [b][kvh][HD][S]
    _Float16* o = vt + ((size_t)(bi * 8 + hh) * 128) * 2048 + si;
    o[(size_t)(d0    ) * 2048] = (_Float16)x0;
    o[(size_t)(d0 + 1) * 2048] = (_Float16)x1;
    o[(size_t)(d0 + 2) * 2048] = (_Float16)x2;
    o[(size_t)(d0 + 3) * 2048] = (_Float16)x3;
    return;
  }
  float ss = x0*x0 + x1*x1 + x2*x2 + x3*x3;
#pragma unroll
  for (int off = 16; off > 0; off >>= 1) ss += __shfl_xor(ss, off, 32);
  const float rn = rsqrtf(ss * (1.0f / 128.0f) + 1e-6f);
  const float* w = kind ? kw : qw;
  x0 *= rn * w[d0]; x1 *= rn * w[d0 + 1];
  x2 *= rn * w[d0 + 2]; x3 *= rn * w[d0 + 3];
  const float Cf = -0.14391157f;                 // -2*ln(10000)/128
  const float f0 = __expf(Cf * (float)(2 * t));
  const float f1 = __expf(Cf * (float)(2 * t + 1));
  const float a0 = (float)si * f0, a1 = (float)si * f1;
  const float c0 = __cosf(a0), s0 = __sinf(a0);
  const float c1 = __cosf(a1), s1 = __sinf(a1);
  const float y0 = x0 * c0 - x1 * s0, y1 = x0 * s0 + x1 * c0;
  const float y2 = x2 * c1 - x3 * s1, y3 = x2 * s1 + x3 * c1;
  _Float16* o = (kind == 0)
      ? qo + ((size_t)(bi * 32 + hh) * 2048 + si) * 128 + d0
      : ko + ((size_t)(bi * 8 + hh) * 2048 + si) * 128 + d0;
  o[0] = (_Float16)y0; o[1] = (_Float16)y1;
  o[2] = (_Float16)y2; o[3] = (_Float16)y3;
}

// ---------------- Flash attention (GQA, causal) ----------------
// grid (qt=32, h=32, b=2), 128 threads = 4 waves, one wave = 16 query rows.
__global__ __launch_bounds__(128) void attn(const _Float16* __restrict__ q,
                                            const _Float16* __restrict__ k,
                                            const _Float16* __restrict__ vt,
                                            _Float16* __restrict__ out) {
  const int S = 2048;
  const int qt = blockIdx.x, h = blockIdx.y, bi = blockIdx.z;
  const int wave = threadIdx.x >> 5, lane = threadIdx.x & 31;
  const int ln = lane & 15, hf = lane >> 4;
  const int qbase = qt * 64 + wave * 16;
  const int kvh = h >> 2;                         // 4 query heads per KV head
  const _Float16* qp = q  + ((size_t)(bi * 32 + h)   * S) * 128;
  const _Float16* kp = k  + ((size_t)(bi * 8 + kvh)  * S) * 128;
  const _Float16* vp = vt + ((size_t)(bi * 8 + kvh)  * 128) * S;

  __shared__ __align__(16) _Float16 sP[4][16][32];  // per-wave P tile

  AF qf[4];
  {
    const _Float16* qr = qp + (size_t)(qbase + ln) * 128;
#pragma unroll
    for (int kc = 0; kc < 4; kc++) {
      const _Float16* p = qr + kc * 32 + hf * 8;
      qf[kc].h[0] = *(const v8h*)p;
      qf[kc].h[1] = *(const v8h*)(p + 16);
    }
  }
  float mrun[8], lrun[8];
  v8f o[8];
#pragma unroll
  for (int r = 0; r < 8; r++) { mrun[r] = -3.0e38f; lrun[r] = 0.0f; }
#pragma unroll
  for (int nt = 0; nt < 8; nt++)
#pragma unroll
    for (int r = 0; r < 8; r++) o[nt][r] = 0.0f;

  const float scale = 0.088388347f;               // 1/sqrt(128)
  const int nkt = (qbase + 47) >> 5;              // key tiles of 32 (causal)
  for (int j = 0; j < nkt; j++) {
    const int kbase = j * 32;
    v8f sf[2];
#pragma unroll
    for (int t = 0; t < 2; t++) {                 // S = Q * K^T (16 x 32 keys)
      v8f sc;
#pragma unroll
      for (int r = 0; r < 8; r++) sc[r] = 0.0f;
#pragma unroll
      for (int kc = 0; kc < 4; kc++) {
        const _Float16* kb =
            kp + (size_t)(kbase + t * 16 + ln) * 128 + kc * 32 + hf * 8;
        AF bf;
        bf.h[0] = *(const v8h*)kb;
        bf.h[1] = *(const v8h*)(kb + 16);
        sc = wmma16(qf[kc].v, bf.v, sc);
      }
      sf[t] = sc;
    }
    const int rq = qbase + 8 * hf;
#pragma unroll
    for (int r = 0; r < 8; r++) {                 // scale + causal mask
      const int qr = rq + r;
      sf[0][r] = (kbase + ln      <= qr) ? sf[0][r] * scale : -3.0e38f;
      sf[1][r] = (kbase + 16 + ln <= qr) ? sf[1][r] * scale : -3.0e38f;
    }
    float vmax[8], rs[8], al[8];
#pragma unroll
    for (int r = 0; r < 8; r++) vmax[r] = fmaxf(sf[0][r], sf[1][r]);
#pragma unroll
    for (int off = 8; off > 0; off >>= 1)
#pragma unroll
      for (int r = 0; r < 8; r++)
        vmax[r] = fmaxf(vmax[r], __shfl_xor(vmax[r], off, 32));
#pragma unroll
    for (int r = 0; r < 8; r++) {
      const float mn = fmaxf(mrun[r], vmax[r]);
      al[r] = __expf(mrun[r] - mn);
      mrun[r] = mn;
      sf[0][r] = __expf(sf[0][r] - mn);
      sf[1][r] = __expf(sf[1][r] - mn);
      rs[r] = sf[0][r] + sf[1][r];
    }
#pragma unroll
    for (int off = 8; off > 0; off >>= 1)
#pragma unroll
      for (int r = 0; r < 8; r++) rs[r] += __shfl_xor(rs[r], off, 32);
#pragma unroll
    for (int r = 0; r < 8; r++) lrun[r] = lrun[r] * al[r] + rs[r];
#pragma unroll
    for (int nt = 0; nt < 8; nt++)
#pragma unroll
      for (int r = 0; r < 8; r++) o[nt][r] *= al[r];
    // P: C-layout -> LDS -> A-layout (per-wave region, DS is in-order per wave)
#pragma unroll
    for (int r = 0; r < 8; r++) {
      sP[wave][r + 8 * hf][ln]      = (_Float16)sf[0][r];
      sP[wave][r + 8 * hf][16 + ln] = (_Float16)sf[1][r];
    }
    AF pf;
    {
      const _Float16* p = &sP[wave][ln][hf * 8];
      pf.h[0] = *(const v8h*)p;
      pf.h[1] = *(const v8h*)(p + 16);
    }
#pragma unroll
    for (int nt = 0; nt < 8; nt++) {              // O += P * V
      const _Float16* vb = vp + (size_t)(nt * 16 + ln) * S + kbase + hf * 8;
      AF vf;
      vf.h[0] = *(const v8h*)vb;
      vf.h[1] = *(const v8h*)(vb + 16);
      o[nt] = wmma16(pf.v, vf.v, o[nt]);
    }
  }
#pragma unroll
  for (int nt = 0; nt < 8; nt++)
#pragma unroll
    for (int r = 0; r < 8; r++) {
      const int row = qbase + r + 8 * hf;
      out[(size_t)(bi * S + row) * 4096 + h * 128 + nt * 16 + ln] =
          (_Float16)(o[nt][r] / lrun[r]);
    }
}

// ---------------- launch ----------------
extern "C" void kernel_launch(void* const* d_in, const int* in_sizes, int n_in,
                              void* d_out, int out_size, void* d_ws, size_t ws_size,
                              hipStream_t stream) {
  const float* x  = (const float*)d_in[0];
  const float* Wq = (const float*)d_in[1];
  const float* Wk = (const float*)d_in[2];
  const float* Wv = (const float*)d_in[3];
  const float* Wo = (const float*)d_in[4];
  const float* qw = (const float*)d_in[5];
  const float* kw = (const float*)d_in[6];
  float* out = (float*)d_out;

  char* w = (char*)d_ws;
  _Float16* x_h    = (_Float16*)w; w += (size_t)4096 * 4096 * 2;
  _Float16* wqkv_h = (_Float16*)w; w += (size_t)4096 * 6144 * 2;
  _Float16* wo_h   = (_Float16*)w; w += (size_t)4096 * 4096 * 2;
  float*    qkv_f  = (float*)w;    w += (size_t)4096 * 6144 * 4;
  _Float16* q_h    = (_Float16*)w; w += (size_t)2 * 32 * 2048 * 128 * 2;
  _Float16* k_h    = (_Float16*)w; w += (size_t)2 * 8 * 2048 * 128 * 2;
  _Float16* vt_h   = (_Float16*)w; w += (size_t)2 * 8 * 2048 * 128 * 2;
  _Float16* attn_h = x_h;          // x_h is dead after the QKV GEMM

  conv1d<<<16384, 256, 0, stream>>>(x,  x_h,  (size_t)4096 * 4096);
  conv2d<<<16384, 256, 0, stream>>>(Wq, wqkv_h, 4096, 4096, 6144, 0);
  conv2d<<< 8192, 256, 0, stream>>>(Wk, wqkv_h, 4096, 1024, 6144, 4096);
  conv2d<<< 8192, 256, 0, stream>>>(Wv, wqkv_h, 4096, 1024, 6144, 5120);
  conv1d<<<16384, 256, 0, stream>>>(Wo, wo_h, (size_t)4096 * 4096);

  gemm_wmma<<<dim3(48, 32), 256, 0, stream>>>(x_h, wqkv_h, qkv_f, 4096, 6144, 4096);
  prep<<<dim3(48, 4096), 32, 0, stream>>>(qkv_f, qw, kw, q_h, k_h, vt_h);
  attn<<<dim3(32, 32, 2), 128, 0, stream>>>(q_h, k_h, vt_h, attn_h);
  gemm_wmma<<<dim3(32, 32), 256, 0, stream>>>(attn_h, wo_h, out, 4096, 4096, 4096);
}
